// MultimodalCausalSelfAttention_59124519797153
// MI455X (gfx1250) — matrix operand order repaired and checked by reading, not compiled, not run on hardware
//
#include <hip/hip_runtime.h>
#include <hip/hip_bf16.h>

typedef __attribute__((ext_vector_type(16))) _Float16 v16h;
typedef __attribute__((ext_vector_type(8)))  _Float16 v8h;
typedef __attribute__((ext_vector_type(8)))  float    v8f;

#define NB   8      // batch
#define NH   8      // heads
#define TOK  1024   // tokens (32*32)
#define CIN  256    // input channels
#define DQK  64     // q/k head dim
#define DV   32     // v head dim

__device__ __forceinline__ v8f wmma_f16(v16h a, v16h b, v8f c) {
  // D = A(16x32 f16) * B(32x16 f16) + C(16x16 f32)
  return __builtin_amdgcn_wmma_f32_16x16x32_f16(
      /*neg_a=*/false, a, /*neg_b=*/false, b,
      /*c_mod=*/(short)0, c, /*reuse_a=*/false, /*reuse_b=*/false);
}

// ---------------------------------------------------------------------------
// Kernel 0a: x (f32, [n][c][t]) -> xh (f16, [n][t][c]) via LDS-tiled transpose.
// Coalesced 128B reads and 64B writes; done once so the GEMM B-operand becomes
// a contiguous 32B f16 load.
// ---------------------------------------------------------------------------
__global__ __launch_bounds__(256)
void x_to_f16_kernel(const float* __restrict__ x, _Float16* __restrict__ xh)
{
  __shared__ _Float16 tile[32][33];
  const int b   = blockIdx.x;            // 8 * 8 * 32 = 2048 blocks
  const int n   = b >> 8;
  const int rem = b & 255;
  const int cb  = rem >> 5;              // channel block (0..7)
  const int tb  = rem & 31;              // token block   (0..31)
  const int tid = threadIdx.x;

#pragma unroll
  for (int i = 0; i < 4; ++i) {
    const int idx = tid + i * 256;
    const int c = idx >> 5, t = idx & 31;   // read row-of-t coalesced
    tile[t][c] = (_Float16)x[((size_t)n * CIN + cb * 32 + c) * TOK + tb * 32 + t];
  }
  __syncthreads();
#pragma unroll
  for (int i = 0; i < 4; ++i) {
    const int idx = tid + i * 256;
    const int t = idx >> 5, c = idx & 31;   // write row-of-c coalesced
    xh[((size_t)n * TOK + tb * 32 + t) * CIN + cb * 32 + c] = tile[t][c];
  }
}

// ---------------------------------------------------------------------------
// Kernel 0b: weights -> f16 once.  Wh = [Wq ; Wkv] (1280 x 256), Woh (256 x 256).
// ---------------------------------------------------------------------------
__global__ __launch_bounds__(256)
void w_to_f16_kernel(const float* __restrict__ Wq, const float* __restrict__ Wkv,
                     const float* __restrict__ Wo,
                     _Float16* __restrict__ Wh, _Float16* __restrict__ Woh)
{
  const int row = blockIdx.x;            // 0..1535
  const int col = threadIdx.x;           // 0..255
  if (row < 512) {
    Wh[(size_t)row * 256 + col] = (_Float16)Wq[(size_t)row * 256 + col];
  } else if (row < 1280) {
    Wh[(size_t)row * 256 + col] = (_Float16)Wkv[(size_t)(row - 512) * 256 + col];
  } else {
    const int r = row - 1280;
    Woh[(size_t)r * 256 + col] = (_Float16)Wo[(size_t)r * 256 + col];
  }
}

// ---------------------------------------------------------------------------
// Kernel 1: fused QKV projection, all-f16 operands.
//   rows 0..511    -> qws [n][h][t][64]  (row-major)
//   rows 512..1023 -> kws [n][h][t][64]  (row-major)
//   rows 1024..1279-> vws [n][h][32][t]  (V^T)
// One wave computes 1 row-tile x 4 token-tiles: the A-tile is loaded once per
// K-chunk and feeds 4 WMMAs (4x arithmetic intensity on the weight fetch).
// ---------------------------------------------------------------------------
__global__ __launch_bounds__(256)
void qkv_proj_kernel(const _Float16* __restrict__ xh,
                     const _Float16* __restrict__ Wh,
                     const float* __restrict__ bq, const float* __restrict__ bkv,
                     _Float16* __restrict__ qws, _Float16* __restrict__ kws,
                     _Float16* __restrict__ vws)
{
  const int wave = threadIdx.x >> 5;
  const int lane = threadIdx.x & 31;
  const int m  = lane & 15;
  const int hi = lane >> 4;

  const int g   = blockIdx.x * 8 + wave;       // 0..10239 (exact)
  const int n   = g / (80 * 16);
  const int rem = g % (80 * 16);
  const int ot  = rem >> 4;                    // output-row tile (0..79)
  const int ttg = rem & 15;                    // token-tile group (4 tiles)

  const _Float16* wrow = Wh + (size_t)(ot * 16 + m) * CIN;   // A row for lane
  const _Float16* xrow[4];
#pragma unroll
  for (int u = 0; u < 4; ++u)
    xrow[u] = xh + ((size_t)n * TOK + (ttg * 4 + u) * 16 + m) * CIN;

  v8f acc[4] = {};
#pragma unroll
  for (int c0 = 0; c0 < CIN; c0 += 32) {
    // A (16x32 f16): elems 0..15 -> K = hi*16+j (contiguous)
    v16h a;
    {
      const v8h a0 = *(const v8h*)(wrow + c0 + hi * 16);
      const v8h a1 = *(const v8h*)(wrow + c0 + hi * 16 + 8);
#pragma unroll
      for (int j = 0; j < 8; ++j) { a[j] = a0[j]; a[j + 8] = a1[j]; }
    }
#pragma unroll
    for (int u = 0; u < 4; ++u) {
      // B (32x16 f16): lane col = token, elem j -> K = hi*16 + j (contiguous)
      const v16h b = *(const v16h*)(xrow[u] + c0 + hi * 16);
      acc[u] = wmma_f16(a, b, acc[u]);
    }
  }

  // C layout: row = v + hi*8, col = m. Region is uniform per wave (ot-based).
#pragma unroll
  for (int u = 0; u < 4; ++u) {
    const int t = (ttg * 4 + u) * 16 + m;
#pragma unroll
    for (int v = 0; v < 8; ++v) {
      const int r = ot * 16 + v + hi * 8;
      const float bias = (r < 512) ? bq[r] : bkv[r - 512];
      const _Float16 val = (_Float16)(acc[u][v] + bias);
      if (r < 512) {
        const int h = r >> 6, d = r & 63;
        qws[(((size_t)n * NH + h) * TOK + t) * DQK + d] = val;
      } else if (r < 1024) {
        const int r2 = r - 512, h = r2 >> 6, d = r2 & 63;
        kws[(((size_t)n * NH + h) * TOK + t) * DQK + d] = val;
      } else {
        const int r2 = r - 1024, h = r2 >> 5, d = r2 & 31;
        vws[(((size_t)n * NH + h) * DV + d) * TOK + t] = val;
      }
    }
  }
}

// ---------------------------------------------------------------------------
// Kernel 2: causal flash attention. One wave per (n, h, 16-row tile).
// S/P never touch memory; P is re-laid-out C->A through a per-wave LDS slab.
// ---------------------------------------------------------------------------
__global__ __launch_bounds__(256)
void attn_kernel(const _Float16* __restrict__ qws, const _Float16* __restrict__ kws,
                 const _Float16* __restrict__ vws, _Float16* __restrict__ ao)
{
  __shared__ _Float16 plds[8][16][32];
  const int wave = threadIdx.x >> 5;
  const int lane = threadIdx.x & 31;
  const int m  = lane & 15;
  const int hi = lane >> 4;

  const int wid = blockIdx.x * 8 + wave;   // 0..4095 (exact)
  const int n   = wid >> 9;
  const int rem = wid & 511;
  const int h   = rem >> 6;
  const int tt  = rem & 63;
  const int t0  = tt * 16;

  const _Float16* qb = qws + ((size_t)n * NH + h) * TOK * DQK;
  const _Float16* kb = kws + ((size_t)n * NH + h) * TOK * DQK;
  const _Float16* vb = vws + ((size_t)n * NH + h) * DV * TOK;

  // Q A-tiles (d = 0..63), row = t0 + m.
  v16h qa0, qa1;
  {
    const _Float16* qr = qb + (size_t)(t0 + m) * DQK;
    const v8h q0 = *(const v8h*)(qr + hi * 16);
    const v8h q1 = *(const v8h*)(qr + hi * 16 + 8);
    const v8h q2 = *(const v8h*)(qr + 32 + hi * 16);
    const v8h q3 = *(const v8h*)(qr + 40 + hi * 16);
#pragma unroll
    for (int j = 0; j < 8; ++j) {
      qa0[j] = q0[j]; qa0[j + 8] = q1[j];
      qa1[j] = q2[j]; qa1[j + 8] = q3[j];
    }
  }

  float rmax[8], rsum[8];
#pragma unroll
  for (int v = 0; v < 8; ++v) { rmax[v] = -3.0e38f; rsum[v] = 0.0f; }
  v8f o0 = {}, o1 = {};
  const float scale = 0.125f;                 // 1/sqrt(64)

  const int nch = (t0 >> 5) + 1;              // causal, wave-uniform
  for (int ch = 0; ch < nch; ++ch) {
    const int s0 = ch * 32;
    if (ch + 1 < nch)
      __builtin_prefetch(kb + (size_t)(s0 + 32 + m) * DQK, 0, 3);

    // K^T B-tiles: col s = s0 + ct*16 + m, elem j -> d = dh*32 + hi*16 + j
    const _Float16* kr0 = kb + (size_t)(s0 + m) * DQK;
    const _Float16* kr1 = kb + (size_t)(s0 + 16 + m) * DQK;
    const v16h k00 = *(const v16h*)(kr0 + hi * 16);
    const v16h k01 = *(const v16h*)(kr0 + 32 + hi * 16);
    const v16h k10 = *(const v16h*)(kr1 + hi * 16);
    const v16h k11 = *(const v16h*)(kr1 + 32 + hi * 16);

    const v8f zero = {};
    v8f sc0 = wmma_f16(qa0, k00, zero); sc0 = wmma_f16(qa1, k01, sc0);
    v8f sc1 = wmma_f16(qa0, k10, zero); sc1 = wmma_f16(qa1, k11, sc1);

    float p0[8], p1[8], red[8];
#pragma unroll
    for (int v = 0; v < 8; ++v) {
      const int trow = t0 + v + hi * 8;
      p0[v] = (s0 + m      <= trow) ? sc0[v] * scale : -1.0e30f;
      p1[v] = (s0 + 16 + m <= trow) ? sc1[v] * scale : -1.0e30f;
      red[v] = fmaxf(p0[v], p1[v]);
    }
#pragma unroll
    for (int msk = 1; msk < 16; msk <<= 1)
#pragma unroll
      for (int v = 0; v < 8; ++v)
        red[v] = fmaxf(red[v], __shfl_xor(red[v], msk, 32));

#pragma unroll
    for (int v = 0; v < 8; ++v) {
      const float nm   = fmaxf(rmax[v], red[v]);
      const float corr = __expf(rmax[v] - nm);
      rmax[v] = nm;
      p0[v] = __expf(p0[v] - nm);
      p1[v] = __expf(p1[v] - nm);
      o0[v] *= corr; o1[v] *= corr;
      rsum[v] *= corr;
      red[v] = p0[v] + p1[v];
    }
#pragma unroll
    for (int msk = 1; msk < 16; msk <<= 1)
#pragma unroll
      for (int v = 0; v < 8; ++v)
        red[v] += __shfl_xor(red[v], msk, 32);
#pragma unroll
    for (int v = 0; v < 8; ++v) rsum[v] += red[v];

    // C-layout -> A-layout via per-wave LDS slab.
#pragma unroll
    for (int v = 0; v < 8; ++v) {
      plds[wave][v + hi * 8][m]      = (_Float16)p0[v];
      plds[wave][v + hi * 8][16 + m] = (_Float16)p1[v];
    }
    asm volatile("s_wait_dscnt 0" ::: "memory");
    v16h pa;
    {
      const v8h pl0 = *(const v8h*)(&plds[wave][m][hi * 16]);
      const v8h pl1 = *(const v8h*)(&plds[wave][m][hi * 16 + 8]);
#pragma unroll
      for (int j = 0; j < 8; ++j) { pa[j] = pl0[j]; pa[j + 8] = pl1[j]; }
    }

    // V B-tiles from V^T [dv][s]
    const v16h vb0 = *(const v16h*)(vb + (size_t)m        * TOK + s0 + hi * 16);
    const v16h vb1 = *(const v16h*)(vb + (size_t)(16 + m) * TOK + s0 + hi * 16);
    o0 = wmma_f16(pa, vb0, o0);
    o1 = wmma_f16(pa, vb1, o1);
  }

#pragma unroll
  for (int v = 0; v < 8; ++v) {
    const float inv = 1.0f / rsum[v];
    const int t = t0 + v + hi * 8;
    _Float16* dst = ao + ((size_t)n * TOK + t) * 256 + h * DV;
    dst[m]      = (_Float16)(o0[v] * inv);
    dst[16 + m] = (_Float16)(o1[v] * inv);
  }
}

// ---------------------------------------------------------------------------
// Kernel 3: out = Wo @ ao^T + bo (fp32 out).  Same 1x4 tile batching: one
// Woh A-tile feeds 4 WMMAs against contiguous f16 ao B-tiles.
// ---------------------------------------------------------------------------
__global__ __launch_bounds__(256)
void out_proj_kernel(const _Float16* __restrict__ ao, const _Float16* __restrict__ Woh,
                     const float* __restrict__ bo, float* __restrict__ out)
{
  const int wave = threadIdx.x >> 5;
  const int lane = threadIdx.x & 31;
  const int m  = lane & 15;
  const int hi = lane >> 4;

  const int g   = blockIdx.x * 8 + wave;   // 0..2047 (exact)
  const int n   = g >> 8;
  const int rem = g & 255;
  const int ot  = rem >> 4;                // 0..15
  const int ttg = rem & 15;                // token-tile group (4 tiles)

  const _Float16* wrow = Woh + (size_t)(ot * 16 + m) * 256;
  const _Float16* arow[4];
#pragma unroll
  for (int u = 0; u < 4; ++u)
    arow[u] = ao + ((size_t)n * TOK + (ttg * 4 + u) * 16 + m) * 256;

  v8f acc[4] = {};
#pragma unroll
  for (int c0 = 0; c0 < 256; c0 += 32) {
    v16h a;
    {
      const v8h a0 = *(const v8h*)(wrow + c0 + hi * 16);
      const v8h a1 = *(const v8h*)(wrow + c0 + hi * 16 + 8);
#pragma unroll
      for (int j = 0; j < 8; ++j) { a[j] = a0[j]; a[j + 8] = a1[j]; }
    }
#pragma unroll
    for (int u = 0; u < 4; ++u) {
      const v16h b = *(const v16h*)(arow[u] + c0 + hi * 16);
      acc[u] = wmma_f16(a, b, acc[u]);
    }
  }
#pragma unroll
  for (int u = 0; u < 4; ++u) {
    const int t = (ttg * 4 + u) * 16 + m;
#pragma unroll
    for (int v = 0; v < 8; ++v) {
      const int o = ot * 16 + v + hi * 8;
      out[((size_t)n * 256 + o) * TOK + t] = acc[u][v] + bo[o];
    }
  }
}

// ---------------------------------------------------------------------------
extern "C" void kernel_launch(void* const* d_in, const int* in_sizes, int n_in,
                              void* d_out, int out_size, void* d_ws, size_t ws_size,
                              hipStream_t stream) {
  (void)in_sizes; (void)n_in; (void)out_size; (void)ws_size;
  const float* x   = (const float*)d_in[0];
  const float* Wq  = (const float*)d_in[1];
  const float* bq  = (const float*)d_in[2];
  const float* Wkv = (const float*)d_in[3];
  const float* bkv = (const float*)d_in[4];
  const float* Wo  = (const float*)d_in[5];
  const float* bo  = (const float*)d_in[6];
  float* out = (float*)d_out;

  char* ws = (char*)d_ws;
  _Float16* qws = (_Float16*)(ws);                        //  8 MiB: q  [n][h][t][64]
  _Float16* kws = (_Float16*)(ws + (size_t)( 8u << 20));  //  8 MiB: k  [n][h][t][64]
  _Float16* vws = (_Float16*)(ws + (size_t)(16u << 20));  //  4 MiB: v^T[n][h][32][t]
  _Float16* ao  = (_Float16*)(ws + (size_t)(20u << 20));  //  4 MiB: ao [n][t][256]
  _Float16* xh  = (_Float16*)(ws + (size_t)(24u << 20));  //  4 MiB: xh [n][t][256]
  _Float16* Wh  = (_Float16*)(ws + (size_t)(28u << 20));  // 640 KiB: [Wq;Wkv] f16
  _Float16* Woh = (_Float16*)(ws + (size_t)(28u << 20) + (size_t)(640u << 10)); // 128 KiB

  x_to_f16_kernel<<<2048, 256, 0, stream>>>(x, xh);
  w_to_f16_kernel<<<1536, 256, 0, stream>>>(Wq, Wkv, Wo, Wh, Woh);
  // 8 n * 80 ot * 16 tt-groups = 10240 waves / 8 per block
  qkv_proj_kernel<<<1280, 256, 0, stream>>>(xh, Wh, bq, bkv, qws, kws, vws);
  // 8 n * 8 h * 64 row-tiles = 4096 waves / 8 per block
  attn_kernel<<<512, 256, 0, stream>>>(qws, kws, vws, ao);
  // 8 n * 16 ot * 16 tt-groups = 2048 waves / 8 per block
  out_proj_kernel<<<256, 256, 0, stream>>>(ao, Woh, bo, out);
}
